// PointNetPlusPlusAttnFusion_New_54709293416841
// MI455X (gfx1250) — matrix operand order repaired and checked by reading, not compiled, not run on hardware
//
#include <hip/hip_runtime.h>
#include <hip/hip_bf16.h>

// -------------------------------------------------------------------------
// PointNet++ forward on gfx1250 (MI455X). All pointwise-MLP layers run as
// bf16 WMMA GEMMs (v_wmma_f32_16x16x32_bf16, 8 WMMAs per wave per K-chunk:
// 2 A-frags x 4 B-frags) with BN affine + ReLU fused into the epilogue.
// Ragged K dims (7, 131) are zero-padded by producers so the GEMM hot loop
// is branch-free and fully vectorized (float4 / b128 staging). Weights are
// pre-packed to bf16 once per launch. Geometry is VALU + LDS.
// Workspace requirement: ~300 MB.
// -------------------------------------------------------------------------

typedef __attribute__((ext_vector_type(16))) __bf16 v16bf;
typedef __attribute__((ext_vector_type(8)))  __bf16 v8bf;
typedef __attribute__((ext_vector_type(8)))  float  v8f;

#define BATCH 16
#define N0    4096
#define S1    512
#define NS1   32
#define S2    128
#define NS2   64

// ========================= Weight pre-pack (fp32 -> bf16, zero-padded K) ====
__global__ __launch_bounds__(256) void pack_weights(
    const float* __restrict__ W, int CIN, int Kpad, int COUT,
    __bf16* __restrict__ out)
{
    int t = blockIdx.x * blockDim.x + threadIdx.x;
    if (t >= COUT * Kpad) return;
    int k = t % Kpad, o = t / Kpad;
    out[t] = (__bf16)((k < CIN) ? W[(size_t)o * CIN + k] : 0.0f);
}

// ========================= WMMA GEMM + BN + ReLU =========================
// Y[r, o] = relu( gamma[o] * (sum_k X[r,k] * Wp[o,k] + b[o]) + beta[o] )
// Preconditions: R % 256 == 0, COUT % 64 == 0, K % 32 == 0, X row stride=ldx.
// Block tile 256x64; 8 waves, wave w owns rows [w*32, w*32+32) x all 64 cols:
// 2 A fragments x 4 B fragments -> 8 WMMAs per K-chunk.
__global__ __launch_bounds__(256) void gemm_bn_relu(
    const float* __restrict__ X, int ldx, int R,
    const __bf16* __restrict__ Wp, int K,
    const float* __restrict__ bias, const float* __restrict__ gamma,
    const float* __restrict__ beta,
    float* __restrict__ Y, int COUT)
{
    __shared__ __bf16 sA[256][32];
    __shared__ __bf16 sW[64][32];

    const int tid    = threadIdx.x;
    const int lane   = tid & 31;
    const int wv     = tid >> 5;          // wave id: 32-row group
    const int mrow   = lane & 15;
    const int laneHi = lane >> 4;
    const int rowBase = blockIdx.x * 256;
    const int colBase = blockIdx.y * 64;

    // staging coordinates (branch-free, fully aligned)
    const int wc = tid >> 2;              // W col 0..63
    const int wk = (tid & 3) * 8;         // W k offset 0/8/16/24

    const float*  xp0 = X  + (size_t)(rowBase + tid) * ldx;   // 1 row / thread
    const __bf16* wp0 = Wp + (size_t)(colBase + wc) * K + wk;

    v8f acc[8] = {};                      // [i*4+j]: A-frag i (0/1), B-frag j

    for (int k0 = 0; k0 < K; k0 += 32) {
        // ---- stage A chunk: 256 rows x 32 k, fp32 -> bf16, one row/thread ----
        const float* xp = xp0 + k0;
        float4 x0 = *(const float4*)(xp + 0);
        float4 x1 = *(const float4*)(xp + 4);
        float4 x2 = *(const float4*)(xp + 8);
        float4 x3 = *(const float4*)(xp + 12);
        float4 x4 = *(const float4*)(xp + 16);
        float4 x5 = *(const float4*)(xp + 20);
        float4 x6 = *(const float4*)(xp + 24);
        float4 x7 = *(const float4*)(xp + 28);
        if (k0 + 32 < K) __builtin_prefetch(xp + 32, 0, 3);   // global_prefetch_b8
        v8bf pa0 = { (__bf16)x0.x, (__bf16)x0.y, (__bf16)x0.z, (__bf16)x0.w,
                     (__bf16)x1.x, (__bf16)x1.y, (__bf16)x1.z, (__bf16)x1.w };
        v8bf pa1 = { (__bf16)x2.x, (__bf16)x2.y, (__bf16)x2.z, (__bf16)x2.w,
                     (__bf16)x3.x, (__bf16)x3.y, (__bf16)x3.z, (__bf16)x3.w };
        v8bf pa2 = { (__bf16)x4.x, (__bf16)x4.y, (__bf16)x4.z, (__bf16)x4.w,
                     (__bf16)x5.x, (__bf16)x5.y, (__bf16)x5.z, (__bf16)x5.w };
        v8bf pa3 = { (__bf16)x6.x, (__bf16)x6.y, (__bf16)x6.z, (__bf16)x6.w,
                     (__bf16)x7.x, (__bf16)x7.y, (__bf16)x7.z, (__bf16)x7.w };
        *(v8bf*)&sA[tid][ 0] = pa0;
        *(v8bf*)&sA[tid][ 8] = pa1;
        *(v8bf*)&sA[tid][16] = pa2;
        *(v8bf*)&sA[tid][24] = pa3;
        // ---- stage W chunk: 64 cols x 32 k, bf16 direct b128 copy ----
        *(v8bf*)&sW[wc][wk] = *(const v8bf*)(wp0 + k0);
        __syncthreads();

        // ---- build fragments per ISA 7.12.2 layouts ----
        v16bf a0, a1, b0, b1, b2, b3;
#pragma unroll
        for (int e = 0; e < 16; ++e) {
            // A 16x32: lane holds row (lane%16); K = (e<8?0:16)+laneHi*8+(e&7)
            int ka = ((e < 8) ? 0 : 16) + laneHi * 8 + (e & 7);
            a0[e] = sA[wv * 32 +  0 + mrow][ka];
            a1[e] = sA[wv * 32 + 16 + mrow][ka];
            // B 32x16: lane holds col (lane%16); K = (e<8?0:16)+laneHi*8+(e&7)
            int kb = ((e < 8) ? 0 : 16) + laneHi * 8 + (e & 7);
            b0[e] = sW[ 0 + mrow][kb];
            b1[e] = sW[16 + mrow][kb];
            b2[e] = sW[32 + mrow][kb];
            b3[e] = sW[48 + mrow][kb];
        }
        acc[0] = __builtin_amdgcn_wmma_f32_16x16x32_bf16(false, a0, false, b0, (short)0, acc[0], false, false);
        acc[1] = __builtin_amdgcn_wmma_f32_16x16x32_bf16(false, a0, false, b1, (short)0, acc[1], false, false);
        acc[2] = __builtin_amdgcn_wmma_f32_16x16x32_bf16(false, a0, false, b2, (short)0, acc[2], false, false);
        acc[3] = __builtin_amdgcn_wmma_f32_16x16x32_bf16(false, a0, false, b3, (short)0, acc[3], false, false);
        acc[4] = __builtin_amdgcn_wmma_f32_16x16x32_bf16(false, a1, false, b0, (short)0, acc[4], false, false);
        acc[5] = __builtin_amdgcn_wmma_f32_16x16x32_bf16(false, a1, false, b1, (short)0, acc[5], false, false);
        acc[6] = __builtin_amdgcn_wmma_f32_16x16x32_bf16(false, a1, false, b2, (short)0, acc[6], false, false);
        acc[7] = __builtin_amdgcn_wmma_f32_16x16x32_bf16(false, a1, false, b3, (short)0, acc[7], false, false);
        __syncthreads();
    }

    // ---- epilogue: C/D layout: VGPR v -> M = v + 8*laneHi, N = lane%16 ----
#pragma unroll
    for (int i = 0; i < 2; ++i) {
        const int row0 = rowBase + wv * 32 + i * 16 + laneHi * 8;
#pragma unroll
        for (int j = 0; j < 4; ++j) {
            int col = colBase + j * 16 + mrow;
            float g = gamma[col], bb = bias[col], bt = beta[col];
#pragma unroll
            for (int v = 0; v < 8; ++v) {
                float o = g * (acc[i * 4 + j][v] + bb) + bt;
                Y[(size_t)(row0 + v) * COUT + col] = o > 0.0f ? o : 0.0f;
            }
        }
    }
}

// ========================= Farthest point sampling =========================
__global__ __launch_bounds__(256) void fps_kernel(
    const float* __restrict__ pts, int stride, int N, int npoint,
    int* __restrict__ outIdx, float* __restrict__ outXyz)
{
    __shared__ float sdist[N0];
    __shared__ float rv[256];
    __shared__ int   ri[256];
    __shared__ int   sFar;

    const int b = blockIdx.x, tid = threadIdx.x;
    const float* base = pts + (size_t)b * N * stride;

    for (int i = tid; i < N; i += 256) sdist[i] = 1e10f;
    if (tid == 0) sFar = 0;
    __syncthreads();

    for (int it = 0; it < npoint; ++it) {
        int far = sFar;
        float cx = base[(size_t)far * stride + 0];
        float cy = base[(size_t)far * stride + 1];
        float cz = base[(size_t)far * stride + 2];
        if (tid == 0) {
            outIdx[b * npoint + it] = far;
            float* o = outXyz + ((size_t)b * npoint + it) * 3;
            o[0] = cx; o[1] = cy; o[2] = cz;
        }
        float best = -1.0f; int bidx = 0;
        for (int i = tid; i < N; i += 256) {
            float dx = base[(size_t)i * stride + 0] - cx;
            float dy = base[(size_t)i * stride + 1] - cy;
            float dz = base[(size_t)i * stride + 2] - cz;
            float d  = dx * dx + dy * dy + dz * dz;
            float nd = fminf(sdist[i], d);
            sdist[i] = nd;
            if (nd > best) { best = nd; bidx = i; }   // i ascending -> first max
        }
        rv[tid] = best; ri[tid] = bidx;
        __syncthreads();
        for (int s = 128; s > 0; s >>= 1) {
            if (tid < s) {
                float ov = rv[tid + s]; int oi = ri[tid + s];
                if (ov > rv[tid] || (ov == rv[tid] && oi < ri[tid])) {
                    rv[tid] = ov; ri[tid] = oi;
                }
            }
            __syncthreads();
        }
        if (tid == 0) sFar = ri[0];
        __syncthreads();
    }
}

// ========================= Ball query =========================
__global__ __launch_bounds__(256) void ball_query_kernel(
    const float* __restrict__ xyz, int stride, int N,
    const float* __restrict__ newXyz, int S, int Bc,
    float r2, int nsample, int* __restrict__ outIdx)
{
    int t = blockIdx.x * blockDim.x + threadIdx.x;
    if (t >= Bc * S) return;
    int b = t / S;
    const float* base = xyz + (size_t)b * N * stride;
    float cx = newXyz[(size_t)t * 3 + 0];
    float cy = newXyz[(size_t)t * 3 + 1];
    float cz = newXyz[(size_t)t * 3 + 2];
    int* out = outIdx + (size_t)t * nsample;
    int cnt = 0, first = 0;
    for (int j = 0; j < N && cnt < nsample; ++j) {
        float dx = base[(size_t)j * stride + 0] - cx;
        float dy = base[(size_t)j * stride + 1] - cy;
        float dz = base[(size_t)j * stride + 2] - cz;
        if (dx * dx + dy * dy + dz * dz <= r2) {
            if (cnt == 0) first = j;
            out[cnt++] = j;
        }
    }
    for (int k = cnt; k < nsample; ++k) out[k] = first;
}

// ========================= Grouping SA1: [dxyz(3), raw(4), 0...] stride 32 ==
__global__ __launch_bounds__(256) void group_sa1(
    const float* __restrict__ xyz, const float* __restrict__ newXyz,
    const int* __restrict__ idx, int S, int ns, int N,
    float* __restrict__ X)
{
    int t = blockIdx.x * blockDim.x + threadIdx.x;
    if (t >= BATCH * S * ns) return;
    int bs = t / ns;
    int b  = bs / S;
    int j  = idx[t];
    const float* p = xyz + ((size_t)b * N + j) * 4;
    const float* c = newXyz + (size_t)bs * 3;
    float* o = X + (size_t)t * 32;
    o[0] = p[0] - c[0]; o[1] = p[1] - c[1]; o[2] = p[2] - c[2];
    o[3] = p[0]; o[4] = p[1]; o[5] = p[2]; o[6] = p[3];
#pragma unroll
    for (int k = 7; k < 32; ++k) o[k] = 0.0f;
}

// ========================= Grouping SA2: [dxyz(3), feats(128), 0...] str 160 =
__global__ __launch_bounds__(256) void group_sa2(
    const float* __restrict__ l1xyz, const float* __restrict__ l2xyz,
    const int* __restrict__ idx, const float* __restrict__ feats,
    int s1, int s2, int ns, int Cpad, float* __restrict__ X)
{
    size_t t = (size_t)blockIdx.x * blockDim.x + threadIdx.x;
    size_t total = (size_t)BATCH * s2 * ns * Cpad;
    if (t >= total) return;
    int c = (int)(t % Cpad);
    size_t row = t / Cpad;
    size_t bs  = row / ns;
    int b = (int)(bs / s2);
    int j = idx[row];
    float v;
    if (c < 3)        v = l1xyz[((size_t)b * s1 + j) * 3 + c] - l2xyz[bs * 3 + c];
    else if (c < 131) v = feats[((size_t)b * s1 + j) * 128 + (c - 3)];
    else              v = 0.0f;
    X[t] = v;
}

// ========================= Max pool over the sample dim =========================
__global__ __launch_bounds__(256) void maxpool_kernel(
    const float* __restrict__ in, int ns, int C, long long BS,
    float* __restrict__ out)
{
    size_t t = (size_t)blockIdx.x * blockDim.x + threadIdx.x;
    if (t >= (size_t)BS * C) return;
    int c = (int)(t % C);
    size_t bs = t / C;
    const float* p = in + (bs * ns) * (size_t)C + c;
    float m = -3.4e38f;
    for (int k = 0; k < ns; ++k) m = fmaxf(m, p[(size_t)k * C]);
    out[t] = m;
}

// ========================= 3-NN (indices + inverse-distance weights) ========
__global__ __launch_bounds__(256) void knn3_kernel(
    const float* __restrict__ xyz1, int stride1, int n1,
    const float* __restrict__ xyz2, int s2, int Bc,
    int* __restrict__ outIdx, float* __restrict__ outW)
{
    int t = blockIdx.x * blockDim.x + threadIdx.x;
    if (t >= Bc * n1) return;
    int b = t / n1;
    const float* p = xyz1 + (size_t)t * stride1;
    float px = p[0], py = p[1], pz = p[2];
    const float* q = xyz2 + (size_t)b * s2 * 3;
    float d0 = 3.4e38f, d1 = 3.4e38f, d2 = 3.4e38f;
    int   i0 = 0, i1 = 0, i2 = 0;
    for (int j = 0; j < s2; ++j) {
        float dx = q[j * 3 + 0] - px;
        float dy = q[j * 3 + 1] - py;
        float dz = q[j * 3 + 2] - pz;
        float d  = dx * dx + dy * dy + dz * dz;
        if (d < d0)      { d2 = d1; i2 = i1; d1 = d0; i1 = i0; d0 = d; i0 = j; }
        else if (d < d1) { d2 = d1; i2 = i1; d1 = d; i1 = j; }
        else if (d < d2) { d2 = d; i2 = j; }
    }
    float w0 = 1.0f / (d0 + 1e-8f);
    float w1 = 1.0f / (d1 + 1e-8f);
    float w2 = 1.0f / (d2 + 1e-8f);
    float s  = w0 + w1 + w2;
    int*   oi = outIdx + (size_t)t * 3;
    float* ow = outW   + (size_t)t * 3;
    oi[0] = i0; oi[1] = i1; oi[2] = i2;
    ow[0] = w0 / s; ow[1] = w1 / s; ow[2] = w2 / s;
}

// ========================= Gather + interpolate (+optional concat) ==========
__global__ __launch_bounds__(256) void gather_interp_kernel(
    const float* __restrict__ pts1, int C1,
    const float* __restrict__ pts2, int Csrc, int C2, int s2,
    const int* __restrict__ idx3, const float* __restrict__ w3,
    long long rows, int n1, float* __restrict__ X)
{
    size_t t = (size_t)blockIdx.x * blockDim.x + threadIdx.x;
    int C = C1 + C2;
    if (t >= (size_t)rows * C) return;
    int c = (int)(t % C);
    size_t r = t / C;
    if (c < C1) {
        X[t] = pts1[r * C1 + c];
    } else {
        int ci = (c - C1) % Csrc;
        int b  = (int)(r / n1);
        const int*   id = idx3 + r * 3;
        const float* w  = w3 + r * 3;
        const float* base = pts2 + (size_t)b * s2 * Csrc;
        X[t] = w[0] * base[(size_t)id[0] * Csrc + ci]
             + w[1] * base[(size_t)id[1] * Csrc + ci]
             + w[2] * base[(size_t)id[2] * Csrc + ci];
    }
}

// ============================== Host driver ==============================
extern "C" void kernel_launch(void* const* d_in, const int* in_sizes, int n_in,
                              void* d_out, int out_size, void* d_ws, size_t ws_size,
                              hipStream_t stream)
{
    const float* xyz = (const float*)d_in[0];
    auto P = [&](int i) { return (const float*)d_in[i]; };
    const int sa1 = 1, sa2 = 13, fp2 = 25, fp1 = 33, fp2c = 45, fp1c = 53;
    (void)n_in; (void)in_sizes; (void)out_size;

    // ---- workspace bump allocator ----
    char* wsb = (char*)d_ws;
    size_t off = 0;
    auto alloc = [&](size_t bytes) -> void* {
        void* p = wsb + off;
        off = (off + bytes + 255) & ~(size_t)255;
        return p;
    };
    int*    fpsIdx1 = (int*)   alloc((size_t)BATCH * S1 * 4);
    int*    fpsIdx2 = (int*)   alloc((size_t)BATCH * S2 * 4);
    float*  l1xyz   = (float*) alloc((size_t)BATCH * S1 * 3 * 4);
    float*  l2xyz   = (float*) alloc((size_t)BATCH * S2 * 3 * 4);
    int*    idx1    = (int*)   alloc((size_t)BATCH * S1 * NS1 * 4);
    int*    idx2    = (int*)   alloc((size_t)BATCH * S2 * NS2 * 4);
    float*  l1pts   = (float*) alloc((size_t)BATCH * S1 * 128 * 4);
    float*  l2pts   = (float*) alloc((size_t)BATCH * S2 * 256 * 4);
    float*  l1back  = (float*) alloc((size_t)BATCH * S1 * 256 * 4);
    float*  l1corr  = (float*) alloc((size_t)BATCH * S1 * 256 * 4);
    int*    knnIdxA = (int*)   alloc((size_t)BATCH * S1 * 3 * 4);
    float*  knnWA   = (float*) alloc((size_t)BATCH * S1 * 3 * 4);
    int*    knnIdxB = (int*)   alloc((size_t)BATCH * N0 * 3 * 4);
    float*  knnWB   = (float*) alloc((size_t)BATCH * N0 * 3 * 4);
    __bf16* wpack   = (__bf16*)alloc((size_t)512 * 640 * 2);     // largest layer
    float*  ping    = (float*) alloc((size_t)262144 * 128 * 4);
    float*  pong    = (float*) alloc((size_t)262144 * 128 * 4);
    (void)ws_size;

    float* out      = (float*)d_out;
    float* out_l0   = out + (size_t)BATCH * N0 * 4;
    float* out_corr = out_l0 + (size_t)BATCH * N0 * 128;

    auto blocks = [](size_t total) { return (unsigned)((total + 255) / 256); };
    // pack weights to bf16 (zero-padded K), then run the aligned WMMA GEMM
    auto gemmL = [&](const float* Xp, int ldx, int R, int CIN, int Kpad,
                     int baseIdx, int layer, float* Yp, int COUT) {
        const float* Wf = P(baseIdx + 4 * layer + 0);
        pack_weights<<<blocks((size_t)COUT * Kpad), 256, 0, stream>>>(
            Wf, CIN, Kpad, COUT, wpack);
        dim3 g((unsigned)(R / 256), (unsigned)(COUT / 64));
        gemm_bn_relu<<<g, 256, 0, stream>>>(
            Xp, ldx, R, wpack, Kpad,
            P(baseIdx + 4 * layer + 1), P(baseIdx + 4 * layer + 2),
            P(baseIdx + 4 * layer + 3), Yp, COUT);
    };

    // ---------------- SA1 ----------------
    fps_kernel<<<BATCH, 256, 0, stream>>>(xyz, 4, N0, S1, fpsIdx1, l1xyz);
    ball_query_kernel<<<blocks((size_t)BATCH * S1), 256, 0, stream>>>(
        xyz, 4, N0, l1xyz, S1, BATCH, 0.04f, NS1, idx1);
    const int R1 = BATCH * S1 * NS1;   // 262144
    group_sa1<<<blocks(R1), 256, 0, stream>>>(xyz, l1xyz, idx1, S1, NS1, N0, ping);
    gemmL(ping, 32, R1, 7,  32, sa1, 0, pong, 64);
    gemmL(pong, 64, R1, 64, 64, sa1, 1, ping, 64);
    gemmL(ping, 64, R1, 64, 64, sa1, 2, pong, 128);
    maxpool_kernel<<<blocks((size_t)BATCH * S1 * 128), 256, 0, stream>>>(
        pong, NS1, 128, (long long)BATCH * S1, l1pts);

    // ---------------- SA2 ----------------
    fps_kernel<<<BATCH, 256, 0, stream>>>(l1xyz, 3, S1, S2, fpsIdx2, l2xyz);
    ball_query_kernel<<<blocks((size_t)BATCH * S2), 256, 0, stream>>>(
        l1xyz, 3, S1, l2xyz, S2, BATCH, 0.16f, NS2, idx2);
    const int R2 = BATCH * S2 * NS2;   // 131072
    group_sa2<<<blocks((size_t)R2 * 160), 256, 0, stream>>>(
        l1xyz, l2xyz, idx2, l1pts, S1, S2, NS2, 160, ping);
    gemmL(ping, 160, R2, 131, 160, sa2, 0, pong, 128);
    gemmL(pong, 128, R2, 128, 128, sa2, 1, ping, 128);
    gemmL(ping, 128, R2, 128, 128, sa2, 2, pong, 256);
    maxpool_kernel<<<blocks((size_t)BATCH * S2 * 256), 256, 0, stream>>>(
        pong, NS2, 256, (long long)BATCH * S2, l2pts);

    // ---------------- FP2 / FP2c share 3-NN geometry ----------------
    knn3_kernel<<<blocks((size_t)BATCH * S1), 256, 0, stream>>>(
        l1xyz, 3, S1, l2xyz, S2, BATCH, knnIdxA, knnWA);
    gather_interp_kernel<<<blocks((size_t)BATCH * S1 * 640), 256, 0, stream>>>(
        l1pts, 128, l2pts, 256, 512, S2, knnIdxA, knnWA,
        (long long)BATCH * S1, S1, ping);
    gemmL(ping, 640, BATCH * S1, 640, 640, fp2, 0, pong, 512);
    gemmL(pong, 512, BATCH * S1, 512, 512, fp2, 1, l1back, 256);

    // ---------------- FP1 / FP1c share 3-NN geometry ----------------
    knn3_kernel<<<blocks((size_t)BATCH * N0), 256, 0, stream>>>(
        xyz, 4, N0, l1xyz, S1, BATCH, knnIdxB, knnWB);
    gather_interp_kernel<<<blocks((size_t)BATCH * N0 * 256), 256, 0, stream>>>(
        nullptr, 0, l1back, 256, 256, S1, knnIdxB, knnWB,
        (long long)BATCH * N0, N0, ping);
    gemmL(ping, 256, BATCH * N0, 256, 256, fp1, 0, pong, 256);
    gemmL(pong, 256, BATCH * N0, 256, 256, fp1, 1, ping, 128);
    gemmL(ping, 128, BATCH * N0, 128, 128, fp1, 2, out_l0, 128);

    // ---------------- correlation head ----------------
    gather_interp_kernel<<<blocks((size_t)BATCH * S1 * 640), 256, 0, stream>>>(
        l1pts, 128, l2pts, 256, 512, S2, knnIdxA, knnWA,
        (long long)BATCH * S1, S1, ping);
    gemmL(ping, 640, BATCH * S1, 640, 640, fp2c, 0, pong, 512);
    gemmL(pong, 512, BATCH * S1, 512, 512, fp2c, 1, l1corr, 256);

    gather_interp_kernel<<<blocks((size_t)BATCH * N0 * 256), 256, 0, stream>>>(
        nullptr, 0, l1corr, 256, 256, S1, knnIdxB, knnWB,
        (long long)BATCH * N0, N0, ping);
    gemmL(ping, 256, BATCH * N0, 256, 256, fp1c, 0, pong, 256);
    gemmL(pong, 256, BATCH * N0, 256, 256, fp1c, 1, ping, 128);
    gemmL(ping, 128, BATCH * N0, 128, 128, fp1c, 2, out_corr, 128);

    // output[0] = xyz passthrough
    hipMemcpyAsync(d_out, xyz, (size_t)BATCH * N0 * 4 * sizeof(float),
                   hipMemcpyDeviceToDevice, stream);
}
